// TransformerDecoderBlock_20426864459856
// MI455X (gfx1250) — matrix operand order repaired and checked by reading, not compiled, not run on hardware
//
#include <hip/hip_runtime.h>
#include <hip/hip_bf16.h>
#include <math.h>

// ---------------------------------------------------------------------------
// CDNA5 (gfx1250) wave32 WMMA transformer decoder block.
// GEMMs: bf16 inputs, f32 accumulation via v_wmma_f32_16x16x32_bf16, with
// double-buffered GLOBAL_LOAD_ASYNC_TO_LDS staging (ASYNCcnt pipeline).
// ---------------------------------------------------------------------------

typedef __attribute__((ext_vector_type(16))) __bf16 v16bf;
typedef __attribute__((ext_vector_type(8)))  float  v8f;

#define DEV __device__ __forceinline__

union FragU { v16bf v; uint4 q[2]; };

// A fragment (16x32 bf16, MxK), A row-major with leading dim ld.
// ISA layout: lane<16 -> M=lane, elems 0-7:K=k0+0..7, elems 8-15:K=k0+16..23
//             lane>=16 -> M=lane-16, K ranges shifted by +8.
DEV v16bf load_a_frag(const __bf16* __restrict__ base, int ld, int row0, int k0) {
  const int lane = threadIdx.x & 31;
  const int hf = lane >> 4, m = lane & 15;
  const __bf16* p = base + (size_t)(row0 + m) * ld + (k0 + hf * 8);
  FragU f;
  f.q[0] = *(const uint4*)(p);        // K: k0+hf*8 + 0..7
  f.q[1] = *(const uint4*)(p + 16);   // K: k0+hf*8 + 16..23
  return f.v;
}

// B fragment (32x16 bf16, KxN) from Bt[N][K] row-major (i.e. B transposed).
// ISA layout: lane<16 -> N=lane, elems 0-7:K=k0+0..7, elems 8-15:K=k0+16..23; lane>=16: K shifted by +8
DEV v16bf load_b_frag(const __bf16* __restrict__ base, int ld, int col0, int k0) {
  const int lane = threadIdx.x & 31;
  const int hf = lane >> 4, n = lane & 15;
  const __bf16* p = base + (size_t)(col0 + n) * ld + (k0 + hf * 8);
  FragU f;
  f.q[0] = *(const uint4*)(p);
  f.q[1] = *(const uint4*)(p + 16);
  return f.v;
}

DEV v8f wmma_bf16(v16bf a, v16bf b, v8f c) {
  return __builtin_amdgcn_wmma_f32_16x16x32_bf16(false, a, false, b, (short)0, c,
                                                 false, false);
}

// Async DMA-style copy of one 16B chunk from global to LDS (gfx1250).
// VDST = per-lane LDS byte offset, VADDR = per-lane 64-bit global address.
DEV void async_ld_b128(unsigned lds_off, const void* gaddr) {
  asm volatile("global_load_async_to_lds_b128 %0, %1, off"
               :: "v"(lds_off), "v"(gaddr)
               : "memory");
}

// Stage a [128 rows x 32 cols] bf16 tile (8KB) from row-major global (ld elems)
// into LDS, same row-major [128][32] layout. 256 threads -> 2 async B128 each.
DEV void stage_tile(const __bf16* __restrict__ g, int ld, int row0, int k0,
                    __bf16* sm) {
  const int t = threadIdx.x;
#pragma unroll
  for (int c = t; c < 512; c += 256) {       // 512 chunks of 16B
    const int r = c >> 2;
    const int co = (c & 3) * 8;              // bf16 elements within row
    const __bf16* gp = g + (size_t)(row0 + r) * ld + k0 + co;
    const unsigned lo = (unsigned)(uintptr_t)(sm + r * 32 + co);
    async_ld_b128(lo, gp);
  }
}

// ---------------------------------------------------------------------------
// Elementwise f32 -> bf16 convert
// ---------------------------------------------------------------------------
__global__ __launch_bounds__(256) void cvt_bf16(const float* __restrict__ in,
                                                __bf16* __restrict__ out, int n) {
  int i = (blockIdx.x * 256 + threadIdx.x) * 4;
  if (i < n) {
    float4 v = *(const float4*)(in + i);
    out[i + 0] = (__bf16)v.x;
    out[i + 1] = (__bf16)v.y;
    out[i + 2] = (__bf16)v.z;
    out[i + 3] = (__bf16)v.w;
  }
}

// ---------------------------------------------------------------------------
// Batched transpose + convert: in[b][R][C] f32 -> out[b][C][R] bf16.
// R, C multiples of 32. Block (32,8).
// ---------------------------------------------------------------------------
__global__ __launch_bounds__(256) void transpose_cvt(const float* __restrict__ in,
                                                     __bf16* __restrict__ out,
                                                     int R, int C) {
  __shared__ float tile[32][33];
  const size_t boff = (size_t)blockIdx.z * R * C;
  const int r0 = blockIdx.y * 32, c0 = blockIdx.x * 32;
  for (int i = threadIdx.y; i < 32; i += 8)
    tile[i][threadIdx.x] = in[boff + (size_t)(r0 + i) * C + c0 + threadIdx.x];
  __syncthreads();
  for (int i = threadIdx.y; i < 32; i += 8)
    out[boff + (size_t)(c0 + i) * R + r0 + threadIdx.x] =
        (__bf16)tile[threadIdx.x][i];
}

// ---------------------------------------------------------------------------
// WMMA GEMM with double-buffered async global->LDS staging.
// C[M,N] = A[M,K](bf16,row-major) * Bt[N,K](bf16)
// 256 threads = 8 waves (4x2), wave tile 32x64, block tile 128x128, K-step 32.
// MODE 0: QKV scatter (q,k row-major [B,H,T,HD]; v transposed [B,H,HD,T])
// MODE 1: f32 out = C + bias + residual
// MODE 2: bf16 out = gelu_exact(C + bias)
// ---------------------------------------------------------------------------
struct GemmP {
  const __bf16* A;
  const __bf16* Bt;
  int M, N, K;
  const float* bias;
  const float* resid;
  float* outF;
  __bf16* outB;
  __bf16 *q, *k, *vt;
};

template <int MODE>
__global__ __launch_bounds__(256) void gemm_wmma(GemmP p) {
  __shared__ __bf16 smA[2][128 * 32];   // 8KB each
  __shared__ __bf16 smB[2][128 * 32];
  const int wave = threadIdx.x >> 5;
  const int wm = wave >> 1, wn = wave & 1;
  const int lane = threadIdx.x & 31;
  const int hf = lane >> 4, n = lane & 15;
  const int rowb = blockIdx.y * 128;
  const int colb = blockIdx.x * 128;

  const v8f zero = {0, 0, 0, 0, 0, 0, 0, 0};
  v8f acc[2][4];
#pragma unroll
  for (int i = 0; i < 2; i++)
#pragma unroll
    for (int j = 0; j < 4; j++) acc[i][j] = zero;

  const int nk = p.K / 32;
  // prologue: stage K-step 0 into buffer 0 (4 async B128 per thread)
  stage_tile(p.A, p.K, rowb, 0, smA[0]);
  stage_tile(p.Bt, p.K, colb, 0, smB[0]);

  for (int i = 0; i < nk; i++) {
    const int cur = i & 1;
    if (i + 1 < nk) {
      // issue next tile's copies, then wait for current tile's 4 copies
      stage_tile(p.A, p.K, rowb, (i + 1) * 32, smA[1 - cur]);
      stage_tile(p.Bt, p.K, colb, (i + 1) * 32, smB[1 - cur]);
      asm volatile("s_wait_asynccnt 4" ::: "memory");
    } else {
      asm volatile("s_wait_asynccnt 0" ::: "memory");
    }
    __syncthreads();  // all waves' copies for buffer `cur` are complete

    const __bf16* sa = smA[cur];
    const __bf16* sb = smB[cur];
    v16bf a0 = load_a_frag(sa, 32, wm * 32 + 0, 0);
    v16bf a1 = load_a_frag(sa, 32, wm * 32 + 16, 0);
#pragma unroll
    for (int j = 0; j < 4; j++) {
      v16bf b = load_b_frag(sb, 32, wn * 64 + j * 16, 0);
      acc[0][j] = wmma_bf16(a0, b, acc[0][j]);
      acc[1][j] = wmma_bf16(a1, b, acc[1][j]);
    }
    // ds reads are consumed (s_wait_dscnt before the WMMAs) before this
    // barrier, so the other buffer is free to be overwritten next iteration.
    __syncthreads();
  }

#pragma unroll
  for (int i = 0; i < 2; i++) {
#pragma unroll
    for (int j = 0; j < 4; j++) {
      const int c0 = colb + wn * 64 + j * 16 + n;  // global column for this lane
#pragma unroll
      for (int r = 0; r < 8; r++) {
        const int gr = rowb + wm * 32 + i * 16 + r + hf * 8;  // global row
        const float v = acc[i][j][r];
        if (MODE == 0) {
          const int which = c0 / 768;
          const int rem = c0 - which * 768;
          const int h = rem >> 6, e = rem & 63;
          const int b = gr >> 11, t = gr & 2047;
          if (which == 0)
            p.q[(((size_t)b * 12 + h) * 2048 + t) * 64 + e] = (__bf16)v;
          else if (which == 1)
            p.k[(((size_t)b * 12 + h) * 2048 + t) * 64 + e] = (__bf16)v;
          else
            p.vt[(((size_t)b * 12 + h) * 64 + e) * 2048 + t] = (__bf16)v;
        } else if (MODE == 1) {
          const size_t idx = (size_t)gr * p.N + c0;
          p.outF[idx] = v + p.bias[c0] + p.resid[idx];
        } else {
          const size_t idx = (size_t)gr * p.N + c0;
          const float y = v + p.bias[c0];
          const float g = 0.5f * y * (1.0f + erff(y * 0.70710678118654752f));
          p.outB[idx] = (__bf16)g;
        }
      }
    }
  }
}

// ---------------------------------------------------------------------------
// Flash-style causal attention. One wave per (b,h, 16-query-row tile).
// q,k: [B,H,T,64] bf16 row-major. vt: [B,H,64,T] bf16. out attnb: [B,T,768] bf16.
// Scale = 768^-0.5 (reference uses full embed dim).
// ---------------------------------------------------------------------------
__global__ __launch_bounds__(256) void attn_kernel(const __bf16* __restrict__ q,
                                                   const __bf16* __restrict__ k,
                                                   const __bf16* __restrict__ vt,
                                                   __bf16* __restrict__ attnb) {
  __shared__ __bf16 smP[8][16 * 32];
  const int wave = threadIdx.x >> 5;
  const int lane = threadIdx.x & 31;
  const int hf = lane >> 4, n = lane & 15;
  const int wid = blockIdx.x * 8 + wave;
  const int qt = wid & 127;         // query tile (of 128 per b,h)
  const int bh = wid >> 7;          // 0..47
  const int b = bh / 12, h = bh % 12;

  const __bf16* qb = q + (size_t)bh * 2048 * 64;
  const __bf16* kb = k + (size_t)bh * 2048 * 64;
  const __bf16* vb = vt + (size_t)bh * 64 * 2048;
  __bf16* myP = smP[wave];

  const v16bf aq0 = load_a_frag(qb, 64, qt * 16, 0);
  const v16bf aq1 = load_a_frag(qb, 64, qt * 16, 32);

  const v8f zero = {0, 0, 0, 0, 0, 0, 0, 0};
  v8f o[4] = {zero, zero, zero, zero};
  float mrow[8], lrow[8];
#pragma unroll
  for (int r = 0; r < 8; r++) { mrow[r] = -INFINITY; lrow[r] = 0.f; }

  const float sc = 0.03608439182435161f;  // 768^-0.5
  const int nblk = (qt * 16 + 15) / 32 + 1;

  for (int sb = 0; sb < nblk; sb++) {
    const int s0 = sb * 32;
    v8f s[2];
#pragma unroll
    for (int j = 0; j < 2; j++) {
      v16bf b0 = load_b_frag(kb, 64, s0 + j * 16, 0);
      v16bf b1 = load_b_frag(kb, 64, s0 + j * 16, 32);
      v8f t = zero;
      t = wmma_bf16(aq0, b0, t);
      t = wmma_bf16(aq1, b1, t);
      s[j] = t;
    }
    // online softmax update (rows r+8*hf live across 16 lanes of this half)
#pragma unroll
    for (int r = 0; r < 8; r++) {
      const int trow = qt * 16 + r + hf * 8;
      float v0 = s[0][r] * sc;
      float v1 = s[1][r] * sc;
      if (s0 + n > trow) v0 = -INFINITY;        // causal mask
      if (s0 + 16 + n > trow) v1 = -INFINITY;
      float bm = fmaxf(v0, v1);
#pragma unroll
      for (int off = 1; off < 16; off <<= 1) bm = fmaxf(bm, __shfl_xor(bm, off, 32));
      const float mnew = fmaxf(mrow[r], bm);
      const float fr = __expf(mrow[r] - mnew);
      const float p0 = __expf(v0 - mnew);
      const float p1 = __expf(v1 - mnew);
      float bs = p0 + p1;
#pragma unroll
      for (int off = 1; off < 16; off <<= 1) bs += __shfl_xor(bs, off, 32);
      lrow[r] = lrow[r] * fr + bs;
      mrow[r] = mnew;
#pragma unroll
      for (int e = 0; e < 4; e++) o[e][r] *= fr;
      const int prow = r + hf * 8;
      myP[prow * 32 + n] = (__bf16)p0;           // C-layout -> LDS row-major
      myP[prow * 32 + 16 + n] = (__bf16)p1;
    }
    __builtin_amdgcn_wave_barrier();
    __asm__ __volatile__("" ::: "memory");
    // reload P as an A fragment (16x32) from LDS; LDS ops are in-order per wave
    {
      const __bf16* pp = &myP[(lane & 15) * 32 + hf * 8];
      FragU f;
      f.q[0] = *(const uint4*)(pp);
      f.q[1] = *(const uint4*)(pp + 16);
      const v16bf ap = f.v;
#pragma unroll
      for (int e = 0; e < 4; e++) {
        v16bf bv = load_b_frag(vb, 2048, e * 16, s0);  // vt[e][s] contiguous in s
        o[e] = wmma_bf16(ap, bv, o[e]);
      }
    }
    __builtin_amdgcn_wave_barrier();
    __asm__ __volatile__("" ::: "memory");
  }

#pragma unroll
  for (int e = 0; e < 4; e++) {
#pragma unroll
    for (int r = 0; r < 8; r++) {
      const int t = qt * 16 + r + hf * 8;
      const float val = o[e][r] * (1.0f / lrow[r]);
      attnb[((size_t)(b * 2048 + t)) * 768 + h * 64 + e * 16 + n] = (__bf16)val;
    }
  }
}

// ---------------------------------------------------------------------------
// LayerNorm: per row of x1 [8192,768] f32 -> h bf16 (one block of 256 per row)
// ---------------------------------------------------------------------------
__global__ __launch_bounds__(256) void ln_kernel(const float* __restrict__ x1,
                                                 const float* __restrict__ g,
                                                 const float* __restrict__ bta,
                                                 __bf16* __restrict__ hb) {
  const int row = blockIdx.x;
  const float* xr = x1 + (size_t)row * 768;
  float s1 = 0.f, s2 = 0.f;
  float vals[3];
#pragma unroll
  for (int i = 0; i < 3; i++) {
    float v = xr[threadIdx.x + i * 256];
    vals[i] = v;
    s1 += v;
    s2 += v * v;
  }
#pragma unroll
  for (int off = 1; off < 32; off <<= 1) {
    s1 += __shfl_xor(s1, off, 32);
    s2 += __shfl_xor(s2, off, 32);
  }
  __shared__ float red[2][8];
  __shared__ float stat[2];
  const int wv = threadIdx.x >> 5;
  if ((threadIdx.x & 31) == 0) { red[0][wv] = s1; red[1][wv] = s2; }
  __syncthreads();
  if (threadIdx.x == 0) {
    float a = 0.f, q = 0.f;
    for (int i = 0; i < 8; i++) { a += red[0][i]; q += red[1][i]; }
    const float mu = a * (1.0f / 768.0f);
    const float var = q * (1.0f / 768.0f) - mu * mu;
    stat[0] = mu;
    stat[1] = rsqrtf(var + 1e-5f);
  }
  __syncthreads();
  const float mu = stat[0], rstd = stat[1];
#pragma unroll
  for (int i = 0; i < 3; i++) {
    const int c = threadIdx.x + i * 256;
    hb[(size_t)row * 768 + c] = (__bf16)((vals[i] - mu) * rstd * g[c] + bta[c]);
  }
}

// ---------------------------------------------------------------------------
// Host-side launch
// ---------------------------------------------------------------------------
extern "C" void kernel_launch(void* const* d_in, const int* in_sizes, int n_in,
                              void* d_out, int out_size, void* d_ws, size_t ws_size,
                              hipStream_t stream) {
  (void)in_sizes; (void)n_in; (void)out_size; (void)ws_size;
  const float* x    = (const float*)d_in[0];
  const float* Wq   = (const float*)d_in[1];
  const float* Wk   = (const float*)d_in[2];
  const float* Wv   = (const float*)d_in[3];
  const float* Wp   = (const float*)d_in[4];
  const float* bp   = (const float*)d_in[5];
  const float* ln_g = (const float*)d_in[6];
  const float* ln_b = (const float*)d_in[7];
  const float* W1   = (const float*)d_in[8];
  const float* b1   = (const float*)d_in[9];
  const float* W2   = (const float*)d_in[10];
  const float* b2   = (const float*)d_in[11];

  const int Bd = 4, T = 2048, D = 768, H = 12, HD = 64, F = 3072;
  const int M = Bd * T;  // 8192

  char* ws = (char*)d_ws;
  size_t off = 0;
  auto alloc = [&](size_t bytes) -> char* {
    char* p = ws + off;
    off += (bytes + 255) & ~size_t(255);
    return p;
  };
  __bf16* xb    = (__bf16*)alloc((size_t)M * D * 2);
  __bf16* wqkvT = (__bf16*)alloc((size_t)3 * H * HD * D * 2);  // [3][H][HD][D]
  __bf16* wpT   = (__bf16*)alloc((size_t)D * D * 2);           // [D][D]
  __bf16* w1T   = (__bf16*)alloc((size_t)F * D * 2);           // [F][D]
  __bf16* w2T   = (__bf16*)alloc((size_t)D * F * 2);           // [D][F]
  __bf16* qbuf  = (__bf16*)alloc((size_t)M * D * 2);           // [B,H,T,HD]
  __bf16* kbuf  = (__bf16*)alloc((size_t)M * D * 2);           // [B,H,T,HD]
  __bf16* vtbuf = (__bf16*)alloc((size_t)M * D * 2);           // [B,H,HD,T]
  __bf16* attnb = (__bf16*)alloc((size_t)M * D * 2);           // [B,T,D]
  float*  x1    = (float*)alloc((size_t)M * D * 4);            // msa + x
  __bf16* hb    = (__bf16*)alloc((size_t)M * D * 2);           // LN output
  __bf16* h1    = (__bf16*)alloc((size_t)M * F * 2);           // gelu(mlp1)

  // 1. convert x -> bf16
  {
    const int n = M * D;
    cvt_bf16<<<dim3(n / (256 * 4)), 256, 0, stream>>>(x, xb, n);
  }
  // 2. weight transposes (f32 -> bf16 transposed)
  transpose_cvt<<<dim3(HD / 32, D / 32, H), dim3(32, 8), 0, stream>>>(
      Wq, wqkvT + (size_t)0 * H * HD * D, D, HD);
  transpose_cvt<<<dim3(HD / 32, D / 32, H), dim3(32, 8), 0, stream>>>(
      Wk, wqkvT + (size_t)1 * H * HD * D, D, HD);
  transpose_cvt<<<dim3(HD / 32, D / 32, H), dim3(32, 8), 0, stream>>>(
      Wv, wqkvT + (size_t)2 * H * HD * D, D, HD);
  transpose_cvt<<<dim3(D / 32, D / 32, 1), dim3(32, 8), 0, stream>>>(Wp, wpT, D, D);
  transpose_cvt<<<dim3(F / 32, D / 32, 1), dim3(32, 8), 0, stream>>>(W1, w1T, D, F);
  transpose_cvt<<<dim3(D / 32, F / 32, 1), dim3(32, 8), 0, stream>>>(W2, w2T, F, D);

  // 3. fused QKV projection: [8192,768] x [768,2304]
  {
    GemmP p{};
    p.A = xb; p.Bt = wqkvT; p.M = M; p.N = 3 * D; p.K = D;
    p.q = qbuf; p.k = kbuf; p.vt = vtbuf;
    gemm_wmma<0><<<dim3((3 * D) / 128, M / 128), 256, 0, stream>>>(p);
  }
  // 4. causal flash attention
  attn_kernel<<<dim3((Bd * H * (T / 16)) / 8), 256, 0, stream>>>(qbuf, kbuf, vtbuf,
                                                                 attnb);
  // 5. output projection + bias + residual(x) -> x1 (f32)
  {
    GemmP p{};
    p.A = attnb; p.Bt = wpT; p.M = M; p.N = D; p.K = D;
    p.bias = bp; p.resid = x; p.outF = x1;
    gemm_wmma<1><<<dim3(D / 128, M / 128), 256, 0, stream>>>(p);
  }
  // 6. LayerNorm -> hb (bf16)
  ln_kernel<<<dim3(M), 256, 0, stream>>>(x1, ln_g, ln_b, hb);
  // 7. MLP1 + bias + exact GELU -> h1 (bf16)
  {
    GemmP p{};
    p.A = hb; p.Bt = w1T; p.M = M; p.N = F; p.K = D;
    p.bias = b1; p.outB = h1;
    gemm_wmma<2><<<dim3(F / 128, M / 128), 256, 0, stream>>>(p);
  }
  // 8. MLP2 + bias + residual(x1) -> out (f32)
  {
    GemmP p{};
    p.A = h1; p.Bt = w2T; p.M = M; p.N = D; p.K = F;
    p.bias = b2; p.resid = x1; p.outF = (float*)d_out;
    gemm_wmma<1><<<dim3(D / 128, M / 128), 256, 0, stream>>>(p);
  }
}
